// GPTBlock_50294067036203
// MI455X (gfx1250) — matrix operand order repaired and checked by reading, not compiled, not run on hardware
//
#include <hip/hip_runtime.h>
#include <hip/hip_bf16.h>

typedef _Float16 f16;
typedef __attribute__((ext_vector_type(16))) _Float16 v16h;
typedef __attribute__((ext_vector_type(8)))  _Float16 v8h;
typedef __attribute__((ext_vector_type(2)))  _Float16 v2h;
typedef __attribute__((ext_vector_type(8)))  float    v8f;

#define B_  4
#define T_  2048
#define D_  512
#define H_  8
#define HD_ 64
#define HALF_ 32
#define MROWS (B_*T_)   // 8192

// ---------------------------------------------------------------------------
// fragment helpers (CDNA5 wave32 WMMA layouts, 16x16x32 f16 -> f32)
// A (16x32, row-major): lane L holds row M=L%16; khalf=L/16; elements are two
// contiguous 8-half chunks at K = khalf*8 and K = khalf*8 + 16.
// B (32x16, col-major access): lane L holds column N=L%16; K in two 8-half chunks
// at K = khalf*8 and K = khalf*8 + 16, as for A.
// C/D f32: VGPR r -> M = r + 8*(L/16), N = L%16.
// ---------------------------------------------------------------------------
__device__ __forceinline__ v16h load_frag_a(const f16* p /* row base + khalf*8 */) {
    v8h lo = *(const v8h*)(p);
    v8h hi = *(const v8h*)(p + 16);
    return __builtin_shufflevector(lo, hi, 0,1,2,3,4,5,6,7,8,9,10,11,12,13,14,15);
}
__device__ __forceinline__ v16h load_frag_b(const f16* p /* col base + khalf*8 */) {
    v8h lo = *(const v8h*)(p);
    v8h hi = *(const v8h*)(p + 16);
    return __builtin_shufflevector(lo, hi, 0,1,2,3,4,5,6,7,8,9,10,11,12,13,14,15);
}
__device__ __forceinline__ v8f wmma_f16(v16h a, v16h b, v8f c) {
    return __builtin_amdgcn_wmma_f32_16x16x32_f16(false, a, false, b, (short)0, c, false, false);
}

// ---------------------------------------------------------------------------
// weight convert + transpose: in f32 [K,N] row-major -> out f16 [N,K]
// ---------------------------------------------------------------------------
__global__ __launch_bounds__(256) void k_wt_f16t(const float* __restrict__ in,
                                                 f16* __restrict__ out,
                                                 int K, int N) {
    size_t idx = (size_t)blockIdx.x * 256 + threadIdx.x;
    size_t total = (size_t)K * N;
    if (idx >= total) return;
    int k = (int)(idx / N);
    int n = (int)(idx % N);
    out[(size_t)n * K + k] = (f16)in[idx];
}

// ---------------------------------------------------------------------------
// LayerNorm: one wave per row of 512, 8 rows per block. f32 in -> f16 out
// ---------------------------------------------------------------------------
__global__ __launch_bounds__(256) void k_layernorm(const float* __restrict__ x,
                                                   const float* __restrict__ g,
                                                   const float* __restrict__ b,
                                                   f16* __restrict__ out) {
    int lane = threadIdx.x & 31;
    int wave = threadIdx.x >> 5;
    int row  = blockIdx.x * 8 + wave;
    const float* xr = x + (size_t)row * D_;
    float xv[16];
    float sum = 0.f, sq = 0.f;
#pragma unroll
    for (int j = 0; j < 16; ++j) {
        float v = xr[lane + j * 32];
        xv[j] = v; sum += v; sq += v * v;
    }
#pragma unroll
    for (int off = 1; off < 32; off <<= 1) {
        sum += __shfl_xor(sum, off, 32);
        sq  += __shfl_xor(sq,  off, 32);
    }
    float mu = sum * (1.f / D_);
    float var = sq * (1.f / D_) - mu * mu;
    float rs = rsqrtf(var + 1e-5f);
    f16* orow = out + (size_t)row * D_;
#pragma unroll
    for (int j = 0; j < 16; ++j) {
        int c = lane + j * 32;
        orow[c] = (f16)((xv[j] - mu) * rs * g[c] + b[c]);
    }
}

// ---------------------------------------------------------------------------
// WMMA GEMM: C[M,N] = A[M,K](f16,row-major) x BT[N,K](f16) with epilogues.
// Block = 8 waves (4 in M x 2 in N); block tile 128x128; wave tile 32x64.
// B tile (128x32 f16) staged in LDS once per block, software pipelined.
// MODE 0: store f16     MODE 1: +bias, exact GELU, store f16
// MODE 2: (+bias)(+residual) store f32
// ---------------------------------------------------------------------------
template <int MODE>
__global__ __launch_bounds__(256) void k_gemm_wmma(const f16* __restrict__ A,
                                                   const f16* __restrict__ BT,
                                                   const float* __restrict__ bias,
                                                   const float* __restrict__ residual,
                                                   float* __restrict__ Cf,
                                                   f16* __restrict__ Ch,
                                                   int M, int N, int K) {
    // 128 cols x 32 k, rows padded to 40 halfs to spread LDS banks
    __shared__ __align__(16) f16 Bs[128 * 40];
    int tid   = threadIdx.x;
    int lane  = tid & 31;
    int wave  = tid >> 5;
    int lm    = lane & 15;
    int khalf = lane >> 4;
    int mw = wave & 3, nw = wave >> 2;
    int m0 = blockIdx.y * 128 + mw * 32;
    int nB = blockIdx.x * 128;

    v8f acc[2][4];
    v8f z = {0.f,0.f,0.f,0.f,0.f,0.f,0.f,0.f};
#pragma unroll
    for (int ms = 0; ms < 2; ++ms)
#pragma unroll
        for (int t = 0; t < 4; ++t) acc[ms][t] = z;

    const f16* Arow0 = A + (size_t)(m0 + lm) * K + khalf * 8;
    const f16* Arow1 = Arow0 + (size_t)16 * K;

    // cooperative B staging: thread -> (col, k-half)
    int scol = tid & 127;
    int skh  = tid >> 7;
    const f16* bsrc = BT + (size_t)(nB + scol) * K + skh * 16;
    f16* bdst = Bs + scol * 40 + skh * 16;

    for (int k0 = 0; k0 < K; k0 += 32) {
        // issue global loads before the barrier (overlap with prior compute)
        v8h b0 = *(const v8h*)(bsrc + k0);
        v8h b1 = *(const v8h*)(bsrc + k0 + 8);
        __syncthreads();                  // previous iteration's LDS reads done
        *(v8h*)(bdst)     = b0;
        *(v8h*)(bdst + 8) = b1;
        __syncthreads();                  // stores visible to all waves

        v16h a0 = load_frag_a(Arow0 + k0);
        v16h a1 = load_frag_a(Arow1 + k0);
        __builtin_prefetch(Arow0 + k0 + 64, 0, 1);
        __builtin_prefetch(Arow1 + k0 + 64, 0, 1);
#pragma unroll
        for (int nt = 0; nt < 4; ++nt) {
            const f16* bp = Bs + (nw * 64 + nt * 16 + lm) * 40 + khalf * 8;
            v16h bf = load_frag_b(bp);
            acc[0][nt] = wmma_f16(a0, bf, acc[0][nt]);
            acc[1][nt] = wmma_f16(a1, bf, acc[1][nt]);
        }
    }

    int rbase = khalf * 8;
#pragma unroll
    for (int ms = 0; ms < 2; ++ms) {
#pragma unroll
        for (int nt = 0; nt < 4; ++nt) {
#pragma unroll
            for (int r = 0; r < 8; ++r) {
                int row = m0 + ms * 16 + rbase + r;
                int col = nB + nw * 64 + nt * 16 + lm;
                size_t idx = (size_t)row * N + col;
                float v = acc[ms][nt][r];
                if (MODE == 0) {
                    Ch[idx] = (f16)v;
                } else if (MODE == 1) {
                    v += bias[col];
                    v = 0.5f * v * (1.f + erff(v * 0.70710678f));
                    Ch[idx] = (f16)v;
                } else {
                    if (bias)     v += bias[col];
                    if (residual) v += residual[idx];
                    Cf[idx] = v;
                }
            }
        }
    }
}

// ---------------------------------------------------------------------------
// RoPE + head split. qkv f16 [B*T, 3D] -> Q,K f16 [B*H, T, 64] (rope'd),
// V f16 transposed [B*H, 64, T] with kv columns INTERLEAVED inside each
// 32-block: position p = 2*(c%16) + (c/16). This makes the attention raw-S
// LDS spill packable into b32 stores; the P x V contraction sees V rows in
// the same permuted order, so the product is unchanged.
// ---------------------------------------------------------------------------
__global__ __launch_bounds__(256) void k_rope_split(const f16* __restrict__ qkv,
                                                    f16* __restrict__ Qh,
                                                    f16* __restrict__ Kh,
                                                    f16* __restrict__ Vt) {
    size_t idx = (size_t)blockIdx.x * 256 + threadIdx.x;   // [0, 2^21)
    int i = (int)(idx & 31);
    int h = (int)((idx >> 5) & 7);
    int t = (int)((idx >> 8) & 2047);
    int b = (int)(idx >> 19);
    int bh = b * H_ + h;
    const f16* row = qkv + ((size_t)b * T_ + t) * (3 * D_);
    int qo = h * HD_;
    float q1 = (float)row[qo + i],            q2 = (float)row[qo + HALF_ + i];
    float k1 = (float)row[D_ + qo + i],       k2 = (float)row[D_ + qo + HALF_ + i];
    float v1 = (float)row[2 * D_ + qo + i],   v2 = (float)row[2 * D_ + qo + HALF_ + i];
    float ang = (float)t * powf(10000.f, -(float)i / (float)HALF_);
    float s, c;
    sincosf(ang, &s, &c);
    size_t qk = ((size_t)bh * T_ + t) * HD_;
    Qh[qk + i]         = (f16)(q1 * c - q2 * s);
    Qh[qk + HALF_ + i] = (f16)(q1 * s + q2 * c);
    Kh[qk + i]         = (f16)(k1 * c - k2 * s);
    Kh[qk + HALF_ + i] = (f16)(k1 * s + k2 * c);
    // interleave kv position within its 32-block
    int tp = (t & ~31) | ((t & 15) << 1) | ((t >> 4) & 1);
    Vt[((size_t)bh * HD_ + i)         * T_ + tp] = (f16)v1;
    Vt[((size_t)bh * HD_ + HALF_ + i) * T_ + tp] = (f16)v2;
}

// ---------------------------------------------------------------------------
// Flash attention. Block = 8 waves x 16 q rows = 128 rows; KV tiles of 64.
// K tile (64x64) and V^T tile (64x64, kv-interleaved) staged in LDS.
// S = (Q/8)K^T (4 C-fragments); raw S packed (column-interleaved) into LDS
// with b32 stores, read back as two A-fragments where each lane owns half a
// row -> row max/sum need one xor-16 shuffle per 64 kv. exp() in registers
// produces both P fragments directly. alpha broadcast: 8 shfl per 64 kv.
// ---------------------------------------------------------------------------
__global__ __launch_bounds__(256) void k_attention(const f16* __restrict__ Qh,
                                                   const f16* __restrict__ Kh,
                                                   const f16* __restrict__ Vt,
                                                   f16* __restrict__ Y) {
    __shared__ __align__(16) f16 Ks[64 * 72];       // [kvpos][dim],  64->72 pad
    __shared__ __align__(16) f16 Vs[64 * 72];       // [dim][kv64],   64->72 pad
    __shared__ __align__(16) f16 Ps[8][16 * 72];    // per-wave raw S (16x64)
    int tid   = threadIdx.x;
    int lane  = tid & 31;
    int wave  = tid >> 5;
    int lm    = lane & 15;
    int khalf = lane >> 4;
    int bh = blockIdx.y;
    int b  = bh >> 3, h = bh & 7;
    int q0 = blockIdx.x * 128 + wave * 16;

    // Q fragments with 1/sqrt(HD) folded in once
    const f16* qp = Qh + ((size_t)bh * T_ + q0 + lm) * HD_;
    v16h qf[2];
    qf[0] = load_frag_a(qp + khalf * 8);
    qf[1] = load_frag_a(qp + 32 + khalf * 8);
#pragma unroll
    for (int i = 0; i < 16; ++i) {
        qf[0][i] = (f16)((float)qf[0][i] * 0.125f);
        qf[1][i] = (f16)((float)qf[1][i] * 0.125f);
    }

    v8f z = {0.f,0.f,0.f,0.f,0.f,0.f,0.f,0.f};
    v8f o[4];
#pragma unroll
    for (int d = 0; d < 4; ++d) o[d] = z;
    // row stats in "row-per-lane" layout: lane L owns row L%16 of the strip
    float mrow = -1e30f, lrow = 0.f;

    // cooperative staging (2 slots per thread for the 64-wide tile)
    int krow = tid >> 3, kof = (tid & 7) * 8;       // K: rows 0..31 (+32)
    const f16* ksrc0 = Kh + ((size_t)bh * T_ + krow) * HD_ + kof;
    const f16* ksrc1 = ksrc0 + (size_t)32 * HD_;
    f16* kdst0 = Ks + krow * 72 + kof;
    f16* kdst1 = kdst0 + 32 * 72;
    int vdim = tid >> 3, vof = (tid & 7) * 8;       // V: dims 0..31 (+32)
    const f16* vsrc0 = Vt + ((size_t)bh * HD_ + vdim) * T_ + vof;
    const f16* vsrc1 = vsrc0 + (size_t)32 * T_;
    f16* vdst0 = Vs + vdim * 72 + vof;
    f16* vdst1 = vdst0 + 32 * 72;

    f16* pl = Ps[wave];
    int rbase = khalf * 8;

    for (int j0 = 0; j0 < T_; j0 += 64) {
        // issue global loads before the barrier to overlap with prior tile
        v8h kr0 = *(const v8h*)(ksrc0 + (size_t)j0 * HD_);
        v8h kr1 = *(const v8h*)(ksrc1 + (size_t)j0 * HD_);
        v8h vr0 = *(const v8h*)(vsrc0 + j0);
        v8h vr1 = *(const v8h*)(vsrc1 + j0);
        __builtin_prefetch(ksrc0 + (size_t)(j0 + 64) * HD_, 0, 1);
        __builtin_prefetch(vsrc0 + j0 + 64, 0, 1);
        __syncthreads();                  // previous tile's LDS reads done
        *(v8h*)kdst0 = kr0;
        *(v8h*)kdst1 = kr1;
        *(v8h*)vdst0 = vr0;
        *(v8h*)vdst1 = vr1;
        __syncthreads();                  // K/V tile visible

        // S = (Q/8) K^T : four 16x16 C-fragments (columns st*16+lm)
        v8f s[4];
#pragma unroll
        for (int st = 0; st < 4; ++st) {
            s[st] = z;
#pragma unroll
            for (int kc = 0; kc < 2; ++kc) {
                const f16* kp = Ks + (st * 16 + lm) * 72 + kc * 32 + khalf * 8;
                s[st] = wmma_f16(qf[kc], load_frag_b(kp), s[st]);
            }
        }
        // raw S -> LDS, column-interleaved: lane's pair packs into one b32
#pragma unroll
        for (int r = 0; r < 8; ++r) {
            int row = rbase + r;
            v2h p01; p01[0] = (f16)s[0][r]; p01[1] = (f16)s[1][r];
            v2h p23; p23[0] = (f16)s[2][r]; p23[1] = (f16)s[3][r];
            *(v2h*)(pl + row * 72 + 2 * lm)      = p01;
            *(v2h*)(pl + row * 72 + 32 + 2 * lm) = p23;
        }
        asm volatile("s_wait_dscnt 0" ::: "memory");
        v16h raw0 = load_frag_a(pl + lm * 72 + khalf * 8);
        v16h raw1 = load_frag_a(pl + lm * 72 + 32 + khalf * 8);

        // row max (lane holds half of row lm's 64 scores; lane^16 the rest)
        float tm[8];
#pragma unroll
        for (int i = 0; i < 8; ++i)
            tm[i] = fmaxf(fmaxf((float)raw0[i], (float)raw0[i + 8]),
                          fmaxf((float)raw1[i], (float)raw1[i + 8]));
#pragma unroll
        for (int i = 0; i < 4; ++i) tm[i] = fmaxf(tm[i], tm[i + 4]);
        float mx = fmaxf(fmaxf(tm[0], tm[1]), fmaxf(tm[2], tm[3]));
        mx = fmaxf(mx, __shfl_xor(mx, 16, 32));        // merge half-rows
        float mnew  = fmaxf(mrow, mx);
        float alpha = __expf(mrow - mnew);
        mrow = mnew;
        // exp in registers -> P fragments + local sum
        v16h pf0, pf1;
        float ts[8];
#pragma unroll
        for (int i = 0; i < 8; ++i) {
            float e0 = __expf((float)raw0[i]     - mnew);
            float e1 = __expf((float)raw0[i + 8] - mnew);
            float e2 = __expf((float)raw1[i]     - mnew);
            float e3 = __expf((float)raw1[i + 8] - mnew);
            pf0[i] = (f16)e0; pf0[i + 8] = (f16)e1;
            pf1[i] = (f16)e2; pf1[i + 8] = (f16)e3;
            ts[i] = (e0 + e1) + (e2 + e3);
        }
#pragma unroll
        for (int i = 0; i < 4; ++i) ts[i] += ts[i + 4];
        float psum = (ts[0] + ts[1]) + (ts[2] + ts[3]);
        psum += __shfl_xor(psum, 16, 32);
        lrow = lrow * alpha + psum;

        // broadcast alpha to C layout and rescale O
#pragma unroll
        for (int r = 0; r < 8; ++r) {
            float a = __shfl(alpha, rbase + r, 32);
#pragma unroll
            for (int d = 0; d < 4; ++d) o[d][r] *= a;
        }
        // O += P x V (V rows already in the interleaved order)
#pragma unroll
        for (int d = 0; d < 4; ++d) {
            const f16* vp = Vs + (d * 16 + lm) * 72 + khalf * 8;
            o[d] = wmma_f16(pf0, load_frag_b(vp),      o[d]);
            o[d] = wmma_f16(pf1, load_frag_b(vp + 32), o[d]);
        }
    }
    // normalize: broadcast 1/l to C layout
    float linv[8];
#pragma unroll
    for (int r = 0; r < 8; ++r) linv[r] = 1.f / __shfl(lrow, rbase + r, 32);
#pragma unroll
    for (int d = 0; d < 4; ++d) {
#pragma unroll
        for (int r = 0; r < 8; ++r) {
            int trow = q0 + rbase + r;
            float v = o[d][r] * linv[r];
            Y[((size_t)b * T_ + trow) * D_ + h * HD_ + d * 16 + lm] = (f16)v;
        }
    }
}

// ---------------------------------------------------------------------------
// Host launcher
// ---------------------------------------------------------------------------
extern "C" void kernel_launch(void* const* d_in, const int* in_sizes, int n_in,
                              void* d_out, int out_size, void* d_ws, size_t ws_size,
                              hipStream_t stream) {
    (void)in_sizes; (void)n_in; (void)out_size; (void)ws_size;
    const float* x     = (const float*)d_in[0];
    const float* ln1_g = (const float*)d_in[1];
    const float* ln1_b = (const float*)d_in[2];
    const float* w_qkv = (const float*)d_in[3];
    const float* w_out = (const float*)d_in[4];
    const float* ln2_g = (const float*)d_in[5];
    const float* ln2_b = (const float*)d_in[6];
    const float* w1    = (const float*)d_in[7];
    const float* b1    = (const float*)d_in[8];
    const float* w2    = (const float*)d_in[9];
    const float* b2    = (const float*)d_in[10];
    float* out = (float*)d_out;

    char* ws = (char*)d_ws;
    size_t off = 0;
    auto alloc = [&](size_t bytes) { char* p = ws + off; off += (bytes + 255) & ~(size_t)255; return p; };
    f16*   wqkvT = (f16*)alloc((size_t)D_ * 3 * D_ * 2);
    f16*   woutT = (f16*)alloc((size_t)D_ * D_ * 2);
    f16*   w1T   = (f16*)alloc((size_t)D_ * 4 * D_ * 2);
    f16*   w2T   = (f16*)alloc((size_t)4 * D_ * D_ * 2);
    f16*   hA    = (f16*)alloc((size_t)MROWS * D_ * 2);
    f16*   qkvh  = (f16*)alloc((size_t)MROWS * 3 * D_ * 2);
    f16*   Qh    = (f16*)alloc((size_t)MROWS * D_ * 2);
    f16*   Kh    = (f16*)alloc((size_t)MROWS * D_ * 2);
    f16*   Vt    = (f16*)alloc((size_t)MROWS * D_ * 2);
    f16*   Yh    = (f16*)alloc((size_t)MROWS * D_ * 2);
    float* x2    = (float*)alloc((size_t)MROWS * D_ * 4);
    f16*   h2    = (f16*)alloc((size_t)MROWS * D_ * 2);
    f16*   a1    = (f16*)alloc((size_t)MROWS * 4 * D_ * 2);

    // weights -> f16 transposed [N,K]
    k_wt_f16t<<<(D_ * 3 * D_ + 255) / 256, 256, 0, stream>>>(w_qkv, wqkvT, D_, 3 * D_);
    k_wt_f16t<<<(D_ * D_ + 255) / 256, 256, 0, stream>>>(w_out, woutT, D_, D_);
    k_wt_f16t<<<(D_ * 4 * D_ + 255) / 256, 256, 0, stream>>>(w1, w1T, D_, 4 * D_);
    k_wt_f16t<<<(4 * D_ * D_ + 255) / 256, 256, 0, stream>>>(w2, w2T, 4 * D_, D_);

    // LN1
    k_layernorm<<<MROWS / 8, 256, 0, stream>>>(x, ln1_g, ln1_b, hA);
    // QKV GEMM (f16 out)
    k_gemm_wmma<0><<<dim3(3 * D_ / 128, MROWS / 128), 256, 0, stream>>>(
        hA, wqkvT, nullptr, nullptr, nullptr, qkvh, MROWS, 3 * D_, D_);
    // RoPE + head split (+ V transpose, kv-interleaved)
    k_rope_split<<<(B_ * T_ * H_ * HALF_) / 256, 256, 0, stream>>>(qkvh, Qh, Kh, Vt);
    // Flash attention
    k_attention<<<dim3(T_ / 128, B_ * H_), 256, 0, stream>>>(Qh, Kh, Vt, Yh);
    // out-proj + residual -> x2 (f32)
    k_gemm_wmma<2><<<dim3(D_ / 128, MROWS / 128), 256, 0, stream>>>(
        Yh, woutT, nullptr, x, x2, nullptr, MROWS, D_, D_);
    // LN2
    k_layernorm<<<MROWS / 8, 256, 0, stream>>>(x2, ln2_g, ln2_b, h2);
    // FC1 + bias + GELU (f16 out)
    k_gemm_wmma<1><<<dim3(4 * D_ / 128, MROWS / 128), 256, 0, stream>>>(
        h2, w1T, b1, nullptr, nullptr, a1, MROWS, 4 * D_, D_);
    // FC2 + bias + residual -> out (f32)
    k_gemm_wmma<2><<<dim3(D_ / 128, MROWS / 128), 256, 0, stream>>>(
        a1, w2T, b2, x2, out, nullptr, MROWS, D_, 4 * D_);
}
